// StyleBlock_36515811950947
// MI455X (gfx1250) — compile-verified, matrix operand order checked by reading, not run on hardware
//
#include <hip/hip_runtime.h>

typedef __attribute__((ext_vector_type(16))) __bf16        v16bf;
typedef __attribute__((ext_vector_type(8)))  float         v8f;
typedef __attribute__((ext_vector_type(4))) unsigned int   v4u;

#define BATCH 8
#define CCH   512
#define HW    4096
#define KTOT  4608      // 9 taps * 512 channels, k' = tap*512 + channel
#define NKIT  144       // KTOT / TK
#define TM    256
#define TN    128
#define TK    32

union FragBF { v16bf v; v4u q[2]; };

__device__ __forceinline__ unsigned short f2bf(float f) {    // exact RNE (cold path)
    unsigned int u = __float_as_uint(f);
    u += 0x7FFFu + ((u >> 16) & 1u);
    return (unsigned short)(u >> 16);
}

// hot path: round-to-nearest (ties up) both floats, pack high halves via V_PERM_B32
__device__ __forceinline__ unsigned int pack2bf(float lo, float hi) {
    unsigned int ul = __float_as_uint(lo) + 0x8000u;
    unsigned int uh = __float_as_uint(hi) + 0x8000u;
    return __builtin_amdgcn_perm(uh, ul, 0x07060302u);  // {uh.b3,uh.b2,ul.b3,ul.b2}
}

// ---------------- Kernel 1: style vector s[b][o] ----------------
__global__ __launch_bounds__(256)
void style_kernel(const float* __restrict__ w, const float* __restrict__ sw,
                  const float* __restrict__ sb, float* __restrict__ s) {
    int g = blockIdx.x * 256 + threadIdx.x;     // 0..4095
    int b = g >> 9;
    int o = g & 511;
    const float cs = 0.04419417382415922f;      // 1/sqrt(512)
    const float* wr = w  + b * CCH;
    const float* sr = sw + o * CCH;
    float acc = 0.f;
    for (int i = 0; i < CCH; ++i) acc += wr[i] * sr[i];
    s[g] = acc * cs + sb[o];
}

// ---- Kernel 2: modulate + demodulate -> bf16 weights in k'=tap*512+ch layout ----
__global__ __launch_bounds__(128)
void modw_kernel(const float* __restrict__ cw, const float* __restrict__ s,
                 unsigned short* __restrict__ wbf) {
    int b = blockIdx.x >> 9;
    int o = blockIdx.x & 511;
    const float cc = 0.014731391274719738f;     // 1/sqrt(512*9)
    const float* cwo  = cw + (size_t)o * KTOT;  // [I][3][3] flat: i*9 + tap
    const float* srow = s + b * CCH;

    float ss = 0.f;
    for (int kp = (int)threadIdx.x; kp < KTOT; kp += 128) {
        int i = kp & 511;                       // channel
        int r = kp >> 9;                        // tap 0..8
        float v = cwo[i * 9 + r] * cc * srow[i];
        ss += v * v;
    }
    __shared__ float red[128];
    red[threadIdx.x] = ss;
    __syncthreads();
    for (int off = 64; off > 0; off >>= 1) {
        if ((int)threadIdx.x < off) red[threadIdx.x] += red[threadIdx.x + off];
        __syncthreads();
    }
    float demod = rsqrtf(red[0] + 1e-8f);

    unsigned short* dst = wbf + ((size_t)b * CCH + o) * KTOT;
    for (int kp = (int)threadIdx.x; kp < KTOT; kp += 128) {
        int i = kp & 511;
        int r = kp >> 9;
        float v = cwo[i * 9 + r] * cc * srow[i] * demod;
        dst[kp] = f2bf(v);                      // permuted, k'-contiguous rows
    }
}

// -------- Kernel 3: implicit-GEMM grouped conv via WMMA bf16 --------
__global__ __launch_bounds__(256)
void conv_kernel(const float* __restrict__ x, const unsigned short* __restrict__ wbf,
                 const float* __restrict__ noise, const float* __restrict__ bias,
                 const float* __restrict__ scale_noise, float* __restrict__ out) {
    __shared__ __align__(16) unsigned short Alds[2][TM][TK];   // 32 KB
    __shared__ __align__(16) unsigned short Blds[2][TN][TK];   // 16 KB

    int blk = blockIdx.x;                 // 512 blocks
    int b   = blk >> 6;                   // 8 batches (64 blocks each)
    int mt  = (blk >> 5) & 1;             // 2 M-tiles of 256
    int nt  = blk & 31;                   // 32 N-tiles of 128

    int p0 = nt * TN;
    const float*          xb = x   + (size_t)b * CCH * HW;
    const unsigned short* wb = wbf + ((size_t)b * CCH + (size_t)mt * TM) * KTOT;

    int lane = threadIdx.x & 31;
    int wv   = threadIdx.x >> 5;          // 8 waves
    int wm   = (wv & 3) * 64;             // wave M offset in tile (4 x 64 = 256)
    int wn   = (wv >> 2) * 64;            // wave N offset in tile (2 x 64 = 128)

    // B-fill assignment: 2 threads per pixel, 16 channels each
    int pl   = threadIdx.x >> 1;          // 0..127
    int half = threadIdx.x & 1;           // 0/1
    int p    = p0 + pl;                   // pixel this thread fills
    int ph   = p >> 6;
    int pw   = p & 63;

    v8f acc[4][4] = {};

    auto fill = [&](int buf, int k0) {
        // ---- A tile: each thread copies one full 64-byte weight row ----
        const v4u* srcA = (const v4u*)(wb + (size_t)threadIdx.x * KTOT + k0);
        v4u* dstA = (v4u*)(&Alds[buf][threadIdx.x][0]);
        dstA[0] = srcA[0];
        dstA[1] = srcA[1];
        dstA[2] = srcA[2];
        dstA[3] = srcA[3];

        // ---- B tile: 16 channels of one tap at this thread's pixel ----
        int kk = k0 >> 9;                 // tap 0..8 (uniform, scalar)
        int i0 = (k0 & 511) + half * 16;  // channel base for this thread
        int kh = kk / 3;                  // scalar division of tiny constant
        int kw = kk - kh * 3;
        int hh = ph + kh - 1;
        int ww = pw + kw - 1;
        bool valid = ((unsigned)hh < 64u) && ((unsigned)ww < 64u);

        float vals[16];
        if (valid) {
            const float* xp = xb + (size_t)i0 * HW + (hh << 6) + ww;
            #pragma unroll
            for (int q = 0; q < 16; ++q) vals[q] = xp[(size_t)q * HW];
        } else {
            #pragma unroll
            for (int q = 0; q < 16; ++q) vals[q] = 0.f;
        }
        unsigned int pk[8];
        #pragma unroll
        for (int q = 0; q < 8; ++q)
            pk[q] = pack2bf(vals[2 * q], vals[2 * q + 1]);
        v4u* dstB = (v4u*)(&Blds[buf][pl][half * 16]);
        dstB[0] = (v4u){pk[0], pk[1], pk[2], pk[3]};
        dstB[1] = (v4u){pk[4], pk[5], pk[6], pk[7]};
    };

    fill(0, 0);
    int cur = 0;
    int koff = (lane >> 4) * 8;           // shorts
    int q0   = koff >> 3;                 // v4u index: 0 or 1
    int lrow = lane & 15;

    for (int it = 0; it < NKIT; ++it) {
        __syncthreads();                  // buf[cur] ready; prev reads done

        // ---- fragment loads first (independent of the upcoming fill) ----
        FragBF afr[4], bfr[4];
        #pragma unroll
        for (int i = 0; i < 4; ++i) {
            const v4u* pA = (const v4u*)(&Alds[cur][wm + i * 16 + lrow][0]);
            afr[i].q[0] = pA[q0];
            afr[i].q[1] = pA[q0 + 2];
        }
        #pragma unroll
        for (int j = 0; j < 4; ++j) {
            const v4u* pB = (const v4u*)(&Blds[cur][wn + j * 16 + lrow][0]);
            bfr[j].q[0] = pB[q0];
            bfr[j].q[1] = pB[q0 + 2];
        }

        // ---- prefetch next tile into the other buffer ----
        if (it + 1 < NKIT) fill(cur ^ 1, (it + 1) * TK);

        #pragma unroll
        for (int i = 0; i < 4; ++i)
            #pragma unroll
            for (int j = 0; j < 4; ++j)
                acc[i][j] = __builtin_amdgcn_wmma_f32_16x16x32_bf16(
                    false, afr[i].v, false, bfr[j].v,
                    (short)0, acc[i][j], false, false);

        cur ^= 1;
    }

    // ---- epilogue: + scale_noise*noise + bias, LeakyReLU(0.2) ----
    float sn = scale_noise[0];
    const float* nb = noise + (size_t)b * HW;
    int mrb  = (lane >> 4) * 8;
    int ncol = lane & 15;
    #pragma unroll
    for (int i = 0; i < 4; ++i) {
        #pragma unroll
        for (int j = 0; j < 4; ++j) {
            int n  = p0 + wn + j * 16 + ncol;
            float nz = sn * nb[n];
            int m0 = mt * TM + wm + i * 16 + mrb;
            float* orow = out + ((size_t)b * CCH + m0) * HW + n;
            #pragma unroll
            for (int r = 0; r < 8; ++r) {
                float v = acc[i][j][r] + nz + bias[m0 + r];
                v = v > 0.f ? v : 0.2f * v;
                orow[(size_t)r * HW] = v;
            }
        }
    }
}

extern "C" void kernel_launch(void* const* d_in, const int* in_sizes, int n_in,
                              void* d_out, int out_size, void* d_ws, size_t ws_size,
                              hipStream_t stream) {
    (void)in_sizes; (void)n_in; (void)out_size; (void)ws_size;
    const float* x     = (const float*)d_in[0];
    const float* w     = (const float*)d_in[1];
    const float* noise = (const float*)d_in[2];
    const float* cw    = (const float*)d_in[3];
    const float* sw    = (const float*)d_in[4];
    const float* sb    = (const float*)d_in[5];
    const float* bias  = (const float*)d_in[6];
    const float* sn    = (const float*)d_in[7];
    float* out = (float*)d_out;

    float*          s   = (float*)d_ws;                               // 16 KB
    unsigned short* wbf = (unsigned short*)((char*)d_ws + 16384);     // ~37.7 MB

    style_kernel<<<16, 256, 0, stream>>>(w, sw, sb, s);
    modw_kernel<<<BATCH * CCH, 128, 0, stream>>>(cw, s, wbf);
    conv_kernel<<<BATCH * 2 * 32, 256, 0, stream>>>(x, wbf, noise, bias, sn, out);
}
